// MultiHeadSelfAttention_18227841204656
// MI455X (gfx1250) — compile-verified, matrix operand order checked by reading, not run on hardware
//
#include <hip/hip_runtime.h>
#include <stdint.h>

// ---------------------------------------------------------------------------
// Multi-head self-attention for MI455X (gfx1250), bf16 WMMA pipeline with
// Tensor-Data-Mover (TDM) double-buffered tile staging.
//   K0: Wo f32 -> bf16 (one-time, enables TDM staging in K3)
//   K1: per-head QKV projection (x @ W^T), f32 -> bf16, layout [n,h,s,d]
//   K2: flash attention (online softmax); K/V tiles via tensor_load_to_lds
//   K3: output projection (x @ Wo^T + bo); tiles via tensor_load_to_lds
// Workspace: Q/K/V bf16 (16 MB each) + attn-out bf16 (16 MB) + Wo bf16 (2 MB)
// ---------------------------------------------------------------------------

typedef __attribute__((ext_vector_type(16))) __bf16    v16bf;
typedef __attribute__((ext_vector_type(8)))  float     v8f;
typedef __attribute__((ext_vector_type(4)))  uint32_t  v4u;
typedef __attribute__((ext_vector_type(8)))  int32_t   v8i;
typedef __attribute__((ext_vector_type(4)))  int32_t   v4i;

#define N_BATCH 4
#define SEQ     2048
#define EMBED   1024
#define HEADS   16
#define HDIM    64

// A-fragment element j -> K index (16-bit A 16x32, ISA 7.12.2)
__device__ __forceinline__ int kidx(int j, int hl) {
  return (j < 8) ? (hl * 8 + j) : (16 + hl * 8 + (j - 8));
}

__device__ __forceinline__ v8f wmma_bf16(v16bf a, v16bf b, v8f c) {
  return __builtin_amdgcn_wmma_f32_16x16x32_bf16(
      /*neg_a=*/false, a, /*neg_b=*/false, b,
      /*c_mod=*/(short)0, c, /*reuse_a=*/false, /*reuse_b=*/false);
}

__device__ __forceinline__ float rowred_max(float v) {
  v = fmaxf(v, __shfl_xor(v, 1, 32));
  v = fmaxf(v, __shfl_xor(v, 2, 32));
  v = fmaxf(v, __shfl_xor(v, 4, 32));
  v = fmaxf(v, __shfl_xor(v, 8, 32));
  return v;
}
__device__ __forceinline__ float rowred_sum(float v) {
  v += __shfl_xor(v, 1, 32);
  v += __shfl_xor(v, 2, 32);
  v += __shfl_xor(v, 4, 32);
  v += __shfl_xor(v, 8, 32);
  return v;
}

// low 32 bits of a flat LDS pointer == offset within the workgroup's LDS
__device__ __forceinline__ uint32_t lds_off(const void* p) {
  return (uint32_t)(uintptr_t)p;
}

// ---------------------------------------------------------------------------
// TDM: issue a 2-D bf16 tile load (tile_d1 rows of tile_d0 elements, global
// row stride d0_stride elements) into LDS at lds_addr, inserting pad_amt+1
// DWORDs of padding after every 2^(pad_int+1) DWORDs (row padding for LDS
// bank-conflict avoidance). Descriptor per CDNA5 ISA.
// Toolchain uses the 6-arg builtin: (g0, g1, g2, g3, g4, cpol).
// ---------------------------------------------------------------------------
__device__ __forceinline__ void tdm_load_2d_bf16(
    uint32_t lds_addr, const void* gptr,
    uint32_t tensor_d0, uint32_t tensor_d1,
    uint32_t tile_d0, uint32_t tile_d1,
    uint32_t d0_stride, uint32_t pad_int, uint32_t pad_amt) {
  uint64_t ga = (uint64_t)(uintptr_t)gptr;
  v4u g0;
  g0[0] = 1u;                                      // count=1 (valid user D#)
  g0[1] = lds_addr;                                // lds_addr
  g0[2] = (uint32_t)ga;                            // global_addr[31:0]
  g0[3] = (uint32_t)(ga >> 32) | (2u << 30);       // global_addr[56:32]|type=2
  v8i g1;
  g1[0] = (int32_t)((1u << 16)                     // data_size = 2 bytes
                    | (1u << 20)                   // pad_enable
                    | (pad_int << 22)              // pad_interval code
                    | (pad_amt << 25));            // pad_amount code
  g1[1] = (int32_t)(tensor_d0 << 16);              // tensor_dim0[15:0]
  g1[2] = (int32_t)((tensor_d0 >> 16) | (tensor_d1 << 16));
  g1[3] = (int32_t)((tensor_d1 >> 16) | (tile_d0 << 16));
  g1[4] = (int32_t)(tile_d1 & 0xffffu);            // tile_dim1 (tile_dim2=0)
  g1[5] = (int32_t)d0_stride;                      // tensor_dim0_stride lo
  g1[6] = 0;
  g1[7] = 0;
  v4i z4 = {0, 0, 0, 0};                           // groups 2/3 unused (2-D)
  v8i z8 = {0, 0, 0, 0, 0, 0, 0, 0};               // group 4 unused
  __builtin_amdgcn_tensor_load_to_lds(g0, g1, z4, z4, z8, 0);
}

// ---------------------------------------------------------------------------
// K0: Wo f32 -> bf16 (row-major copy)
// ---------------------------------------------------------------------------
__global__ __launch_bounds__(256) void wo_bf16_kernel(
    const float* __restrict__ Wo, __bf16* __restrict__ Wob) {
  int i = (blockIdx.x * 256 + threadIdx.x) * 4;    // grid covers 1024*1024
  float4 v = *(const float4*)(Wo + i);
  Wob[i + 0] = (__bf16)v.x;  Wob[i + 1] = (__bf16)v.y;
  Wob[i + 2] = (__bf16)v.z;  Wob[i + 3] = (__bf16)v.w;
}

// ---------------------------------------------------------------------------
// K1: per-head projection. out[nh][s][e] = sum_d x[n][s][h*64+d] * W[e][d]
// grid (S/128, N*H, 3), 256 threads (8 waves). Each wave: 16 rows x 64 cols.
// ---------------------------------------------------------------------------
__global__ __launch_bounds__(256) void qkv_proj_kernel(
    const float* __restrict__ Xq, const float* __restrict__ Xk,
    const float* __restrict__ Xv,
    const float* __restrict__ Wq, const float* __restrict__ Wk,
    const float* __restrict__ Wv,
    __bf16* __restrict__ Qp, __bf16* __restrict__ Kp, __bf16* __restrict__ Vp) {
  const int z = blockIdx.z;
  const float* __restrict__ X = (z == 0) ? Xq : (z == 1) ? Xk : Xv;
  const float* __restrict__ W = (z == 0) ? Wq : (z == 1) ? Wk : Wv;
  __bf16* __restrict__ Out    = (z == 0) ? Qp : (z == 1) ? Kp : Vp;

  const int nh = blockIdx.y;              // n*HEADS + h
  const int n  = nh >> 4, h = nh & 15;
  const int s0 = blockIdx.x * 128;
  const int tid = threadIdx.x;

  __shared__ __bf16 Xs[128][72];
  __shared__ __bf16 Ws[64][72];

  {  // stage 128x64 input slice + 64x64 W, converting f32 -> bf16
    const float* base = X + ((size_t)n * SEQ + s0) * EMBED + h * HDIM;
#pragma unroll
    for (int i = 0; i < 8; ++i) {
      int fi = tid + i * 256;             // float4 granules, 0..2047
      int row = fi >> 4, c4 = (fi & 15) * 4;
      float4 v = *(const float4*)(base + (size_t)row * EMBED + c4);
      Xs[row][c4 + 0] = (__bf16)v.x;  Xs[row][c4 + 1] = (__bf16)v.y;
      Xs[row][c4 + 2] = (__bf16)v.z;  Xs[row][c4 + 3] = (__bf16)v.w;
    }
#pragma unroll
    for (int i = 0; i < 4; ++i) {
      int fi = tid + i * 256;             // 0..1023
      int row = fi >> 4, c4 = (fi & 15) * 4;
      float4 v = *(const float4*)(W + row * 64 + c4);
      Ws[row][c4 + 0] = (__bf16)v.x;  Ws[row][c4 + 1] = (__bf16)v.y;
      Ws[row][c4 + 2] = (__bf16)v.z;  Ws[row][c4 + 3] = (__bf16)v.w;
    }
  }
  __syncthreads();

  const int wave = tid >> 5, lane = tid & 31;
  const int hl = lane >> 4, lm = lane & 15;

  v16bf a[2];
#pragma unroll
  for (int kb = 0; kb < 2; ++kb)
#pragma unroll
    for (int j = 0; j < 16; ++j)
      a[kb][j] = Xs[wave * 16 + lm][kb * 32 + kidx(j, hl)];

  v8f acc[4] = {};
#pragma unroll
  for (int nt = 0; nt < 4; ++nt)
#pragma unroll
    for (int kb = 0; kb < 2; ++kb) {
      v16bf b;                             // B[k=d][n=e] = W[e][d]
#pragma unroll
      for (int j = 0; j < 16; ++j)
        b[j] = Ws[nt * 16 + lm][kb * 32 + kidx(j, hl)];
      acc[nt] = wmma_bf16(a[kb], b, acc[nt]);
    }

#pragma unroll
  for (int nt = 0; nt < 4; ++nt)
#pragma unroll
    for (int j = 0; j < 8; ++j) {
      int srow = s0 + wave * 16 + j + hl * 8;   // C layout: M = j + half*8
      Out[((size_t)nh * SEQ + srow) * HDIM + nt * 16 + lm] = (__bf16)acc[nt][j];
    }
}

// ---------------------------------------------------------------------------
// K2: flash attention per (n,h). grid (S/128, N*H), 256 threads.
// Wave owns 16 q-rows; 32-key tiles with online softmax. K/V tiles are
// double-buffered in LDS and fetched by wave 0 via TDM (tensor_load_to_lds,
// pad feature produces the 72-element row stride), overlapped with compute.
// ---------------------------------------------------------------------------
__global__ __launch_bounds__(256) void attention_kernel(
    const __bf16* __restrict__ Qp, const __bf16* __restrict__ Kp,
    const __bf16* __restrict__ Vp, __bf16* __restrict__ Ao) {
  const int nh = blockIdx.y;
  const int n = nh >> 4, h = nh & 15;
  const int q0 = blockIdx.x * 128;
  const int tid = threadIdx.x;
  const int wave = tid >> 5, lane = tid & 31;
  const int hl = lane >> 4, lm = lane & 15;
  const bool tdm_wave = (tid < 32);

  __shared__ __bf16 Ks[2][32][72];
  __shared__ __bf16 Vs[2][32][72];
  __shared__ __bf16 Ps[8][16][40];         // per-wave P transpose staging

  const __bf16* Qbase = Qp + (size_t)nh * SEQ * HDIM;
  const __bf16* Kbase = Kp + (size_t)nh * SEQ * HDIM;
  const __bf16* Vbase = Vp + (size_t)nh * SEQ * HDIM;

  // resident Q fragments (16 rows x 64 k, two 16x32 A-frags)
  v16bf qa[2];
  {
    const __bf16* qrow = Qbase + (size_t)(q0 + wave * 16 + lm) * HDIM;
#pragma unroll
    for (int kb = 0; kb < 2; ++kb)
#pragma unroll
      for (int j = 0; j < 16; ++j)
        qa[kb][j] = qrow[kb * 32 + kidx(j, hl)];
  }

  // prologue: TDM-fetch tile 0 into buffer 0 (row 64 bf16 = 32 DWORDs ->
  // pad_interval code 4; pad 4 DWORDs -> code 3 => LDS row stride 72 bf16)
  if (tdm_wave) {
    tdm_load_2d_bf16(lds_off(&Ks[0][0][0]), Kbase, 64, 32, 64, 32, 64, 4, 3);
    tdm_load_2d_bf16(lds_off(&Vs[0][0][0]), Vbase, 64, 32, 64, 32, 64, 4, 3);
  }

  v8f acc[4] = {};
  float rm[8], rl[8];
#pragma unroll
  for (int j = 0; j < 8; ++j) { rm[j] = -__builtin_inff(); rl[j] = 0.f; }

  const float scale = 0.03125f;            // 1/sqrt(EMBED)

  for (int it = 0; it < SEQ / 32; ++it) {
    const int b = it & 1;
    if (tdm_wave) __builtin_amdgcn_s_wait_tensorcnt(0);
    __syncthreads();                       // publish TDM writes; all waves
                                           // done with buffer b^1
    if (it + 1 < SEQ / 32 && tdm_wave) {   // prefetch next tile pair
      const __bf16* kn = Kbase + (size_t)(it + 1) * 32 * HDIM;
      const __bf16* vn = Vbase + (size_t)(it + 1) * 32 * HDIM;
      tdm_load_2d_bf16(lds_off(&Ks[b ^ 1][0][0]), kn, 64, 32, 64, 32, 64, 4, 3);
      tdm_load_2d_bf16(lds_off(&Vs[b ^ 1][0][0]), vn, 64, 32, 64, 32, 64, 4, 3);
    }

    // scores S = Q @ K^T for 16 q-rows x 32 keys (two 16-col groups)
    v8f s[2] = {};
#pragma unroll
    for (int cg = 0; cg < 2; ++cg)
#pragma unroll
      for (int kb = 0; kb < 2; ++kb) {
        v16bf bb;                          // B[k=d][n=key] = K[key][d]
#pragma unroll
        for (int j = 0; j < 16; ++j)
          bb[j] = Ks[b][cg * 16 + lm][kb * 32 + kidx(j, hl)];
        s[cg] = wmma_bf16(qa[kb], bb, s[cg]);
      }

    // online softmax; element j = row (j + half*8), col = lane%16
#pragma unroll
    for (int j = 0; j < 8; ++j) {
      float v0 = s[0][j] * scale, v1 = s[1][j] * scale;
      float mnew = fmaxf(rm[j], rowred_max(fmaxf(v0, v1)));
      float p0 = __expf(v0 - mnew);
      float p1 = __expf(v1 - mnew);
      float corr = __expf(rm[j] - mnew);
      rl[j] = rl[j] * corr + rowred_sum(p0 + p1);
      rm[j] = mnew;
#pragma unroll
      for (int t = 0; t < 4; ++t) acc[t][j] *= corr;
      Ps[wave][j + hl * 8][lm]      = (__bf16)p0;   // C-layout -> LDS
      Ps[wave][j + hl * 8][16 + lm] = (__bf16)p1;
    }

    // reload P in A-layout (wave-local LDS, hw keeps DS in order) and do PV
    v16bf pa;
#pragma unroll
    for (int j = 0; j < 16; ++j)
      pa[j] = Ps[wave][lm][kidx(j, hl)];
#pragma unroll
    for (int ng = 0; ng < 4; ++ng) {
      v16bf vb;                            // B[k=key][n=d] = V[key][d]
#pragma unroll
      for (int j = 0; j < 16; ++j)
        vb[j] = Vs[b][kidx(j, hl)][ng * 16 + lm];
      acc[ng] = wmma_bf16(pa, vb, acc[ng]);
    }
  }

  // normalize, write attention output in [N,S,E] (merged-head) bf16
#pragma unroll
  for (int j = 0; j < 8; ++j) {
    float inv = 1.f / rl[j];
    int srow = q0 + wave * 16 + j + hl * 8;
    size_t rowbase = ((size_t)n * SEQ + srow) * EMBED + h * HDIM;
#pragma unroll
    for (int ng = 0; ng < 4; ++ng)
      Ao[rowbase + ng * 16 + lm] = (__bf16)(acc[ng][j] * inv);
  }
}

// ---------------------------------------------------------------------------
// K3: output projection Out = Ao @ Wob^T + bo  (8192 x 1024 x 1024, bf16).
// grid (EMBED/64, N*S/128), 256 threads; block tile 128x64, k-step 32.
// Both the activation tile (128x32) and the Wo tile (64x32) are TDM-fetched
// into double-buffered LDS (row 32 bf16 = 16 DWORDs -> pad code 3; +4 DWORDs
// pad -> 40-element row stride).
// ---------------------------------------------------------------------------
__global__ __launch_bounds__(256) void out_proj_kernel(
    const __bf16* __restrict__ Ao, const __bf16* __restrict__ Wob,
    const float* __restrict__ bo, float* __restrict__ Out) {
  const int n0 = blockIdx.x * 64;
  const int m0 = blockIdx.y * 128;
  const int tid = threadIdx.x;
  const int wave = tid >> 5, lane = tid & 31;
  const int hl = lane >> 4, lm = lane & 15;
  const bool tdm_wave = (tid < 32);

  __shared__ __bf16 Xs[2][128][40];
  __shared__ __bf16 Ws[2][64][40];         // row-major [n][k] slice of Wob

  const __bf16* Xbase = Ao + (size_t)m0 * EMBED;
  const __bf16* Wbase = Wob + (size_t)n0 * EMBED;

  if (tdm_wave) {
    tdm_load_2d_bf16(lds_off(&Xs[0][0][0]), Xbase, 32, 128, 32, 128, EMBED, 3, 3);
    tdm_load_2d_bf16(lds_off(&Ws[0][0][0]), Wbase, 32, 64, 32, 64, EMBED, 3, 3);
  }

  v8f acc[4] = {};

  for (int it = 0; it < EMBED / 32; ++it) {
    const int b = it & 1;
    if (tdm_wave) __builtin_amdgcn_s_wait_tensorcnt(0);
    __syncthreads();
    if (it + 1 < EMBED / 32 && tdm_wave) {
      const int k0n = (it + 1) * 32;
      tdm_load_2d_bf16(lds_off(&Xs[b ^ 1][0][0]), Xbase + k0n,
                       32, 128, 32, 128, EMBED, 3, 3);
      tdm_load_2d_bf16(lds_off(&Ws[b ^ 1][0][0]), Wbase + k0n,
                       32, 64, 32, 64, EMBED, 3, 3);
    }

    v16bf a;
#pragma unroll
    for (int j = 0; j < 16; ++j)
      a[j] = Xs[b][wave * 16 + lm][kidx(j, hl)];
#pragma unroll
    for (int ng = 0; ng < 4; ++ng) {
      v16bf bb;                            // B[k][n] = Wob[n][k]
#pragma unroll
      for (int j = 0; j < 16; ++j)
        bb[j] = Ws[b][ng * 16 + lm][kidx(j, hl)];
      acc[ng] = wmma_bf16(a, bb, acc[ng]);
    }
  }

#pragma unroll
  for (int ng = 0; ng < 4; ++ng) {
    float bias = bo[n0 + ng * 16 + lm];
#pragma unroll
    for (int j = 0; j < 8; ++j) {
      int row = m0 + wave * 16 + j + hl * 8;
      Out[(size_t)row * EMBED + n0 + ng * 16 + lm] = acc[ng][j] + bias;
    }
  }
}

// ---------------------------------------------------------------------------
extern "C" void kernel_launch(void* const* d_in, const int* in_sizes, int n_in,
                              void* d_out, int out_size, void* d_ws, size_t ws_size,
                              hipStream_t stream) {
  (void)in_sizes; (void)n_in; (void)out_size; (void)ws_size;
  // setup_inputs order: values, keys, query, Wv, Wk, Wq, Wo, bo (all f32)
  const float* values = (const float*)d_in[0];
  const float* keys_  = (const float*)d_in[1];
  const float* query  = (const float*)d_in[2];
  const float* Wv = (const float*)d_in[3];
  const float* Wk = (const float*)d_in[4];
  const float* Wq = (const float*)d_in[5];
  const float* Wo = (const float*)d_in[6];
  const float* bo = (const float*)d_in[7];
  float* out = (float*)d_out;

  const size_t per = (size_t)N_BATCH * HEADS * SEQ * HDIM;  // 8,388,608 elems
  __bf16* Qp  = (__bf16*)d_ws;       // 16 MB
  __bf16* Kp  = Qp + per;            // 16 MB
  __bf16* Vp  = Kp + per;            // 16 MB
  __bf16* Ao  = Vp + per;            // 16 MB
  __bf16* Wob = Ao + per;            // 2 MB   (total 66 MB of d_ws)

  dim3 blk(256);
  wo_bf16_kernel<<<dim3((EMBED * EMBED) / 1024), blk, 0, stream>>>(Wo, Wob);
  qkv_proj_kernel<<<dim3(SEQ / 128, N_BATCH * HEADS, 3), blk, 0, stream>>>(
      query, keys_, values, Wq, Wk, Wv, Qp, Kp, Vp);
  attention_kernel<<<dim3(SEQ / 128, N_BATCH * HEADS), blk, 0, stream>>>(
      Qp, Kp, Vp, Ao);
  out_proj_kernel<<<dim3(EMBED / 64, (N_BATCH * SEQ) / 128), blk, 0, stream>>>(
      Ao, Wob, bo, out);
}
